// MoE_30416958390574
// MI455X (gfx1250) — compile-verified, matrix operand order checked by reading, not run on hardware
//
#include <hip/hip_runtime.h>
#include <hip/hip_bf16.h>

// ---------------------------------------------------------------------------
// MoE: T=8192 tokens, D=1024, E=8 experts, top-2 routing, fp32 in/out.
// Gathered grouped-GEMM, bf16 hi/lo split (3x V_WMMA_F32_16X16X32_BF16 per
// 16x16 tile), double-buffered LDS, one barrier per K-step.
// ---------------------------------------------------------------------------

#define T_TOK 8192
#define DIM   1024
#define NEXP  8

#define MT 32      // M tile (tokens) per block
#define NT 128     // N tile (output cols) per block
#define KT 32      // K step (bf16 WMMA K)
#define LDK 40     // padded LDS row (bf16 elems): 80B -> conflict-free, 16B aligned
#define NSTEP (DIM / KT)

typedef __bf16 bf16;
typedef __attribute__((ext_vector_type(4)))  __bf16 v4bf;
typedef __attribute__((ext_vector_type(8)))  __bf16 v8bf;
typedef __attribute__((ext_vector_type(16))) __bf16 v16bf;
typedef __attribute__((ext_vector_type(8)))  float  v8f;

__device__ __forceinline__ void f32_split(float v, bf16& hi, bf16& lo) {
    hi = (bf16)v;
    lo = (bf16)(v - (float)hi);
}

// ---------------------------------------------------------------------------
// Phase 1: gating. One wave per token: logits[8], top-2, softmax, build
// rank-0 and rank-1 per-expert gather lists via int atomics.
// ---------------------------------------------------------------------------
__global__ __launch_bounds__(256)
void moe_gate(const float* __restrict__ x, const float* __restrict__ gw,
              int* __restrict__ c0, int* __restrict__ c1,
              int* __restrict__ l0, float* __restrict__ w0a,
              int* __restrict__ l1, float* __restrict__ w1a)
{
    const int wave = threadIdx.x >> 5;
    const int lane = threadIdx.x & 31;
    const int t    = blockIdx.x * 8 + wave;

    const float* xr = x + (size_t)t * DIM;
    float acc[NEXP];
#pragma unroll
    for (int e = 0; e < NEXP; ++e) acc[e] = 0.0f;

    for (int d = lane; d < DIM; d += 32) {
        const float xv = xr[d];
#pragma unroll
        for (int e = 0; e < NEXP; ++e)
            acc[e] = fmaf(xv, gw[e * DIM + d], acc[e]);
    }
#pragma unroll
    for (int e = 0; e < NEXP; ++e) {
#pragma unroll
        for (int off = 16; off > 0; off >>= 1)
            acc[e] += __shfl_xor(acc[e], off, 32);
    }

    if (lane == 0) {
        int i0 = 0; float m0v = acc[0];
#pragma unroll
        for (int e = 1; e < NEXP; ++e)
            if (acc[e] > m0v) { m0v = acc[e]; i0 = e; }
        int i1 = -1; float m1v = -3.4e38f;
#pragma unroll
        for (int e = 0; e < NEXP; ++e)
            if (e != i0 && acc[e] > m1v) { m1v = acc[e]; i1 = e; }
        const float a  = expf(m1v - m0v);
        const float dn = 1.0f + a;
        const float wA = 1.0f / dn;
        const float wB = a / dn;

        const int p0 = atomicAdd(&c0[i0], 1);
        l0[i0 * T_TOK + p0]  = t;
        w0a[i0 * T_TOK + p0] = wA;
        const int p1 = atomicAdd(&c1[i1], 1);
        l1[i1 * T_TOK + p1]  = t;
        w1a[i1 * T_TOK + p1] = wB;
    }
}

// ---------------------------------------------------------------------------
// Phase 2: gathered grouped GEMM.  out[tok, nb:nb+128] (+)= w * x[tok] @ We^T
// 256 threads = 8 waves in a 2(M) x 4(N) wave grid; each wave owns a 16x32
// output (2 WMMA tiles).  Double-buffered LDS, 1 barrier / K-step.
// ---------------------------------------------------------------------------
template <int ACC>
__global__ __launch_bounds__(256)
void moe_gemm(const float* __restrict__ x, const float* __restrict__ ew,
              const int* __restrict__ counts, const int* __restrict__ lists,
              const float* __restrict__ wts, float* __restrict__ out)
{
    __shared__ __align__(16) bf16 Ahi[2][MT * LDK];
    __shared__ __align__(16) bf16 Alo[2][MT * LDK];
    __shared__ __align__(16) bf16 Bhi[2][NT * LDK];
    __shared__ __align__(16) bf16 Blo[2][NT * LDK];
    __shared__ int   toks[MT];
    __shared__ float wrow[MT];

    const int e   = blockIdx.z;
    const int cnt = counts[e];
    const int m0  = blockIdx.y * MT;
    if (m0 >= cnt) return;                       // uniform across block
    const int nb  = blockIdx.x * NT;
    const int tid = threadIdx.x;

    if (tid < MT) {
        const int r  = m0 + tid;
        const bool ok = (r < cnt);
        toks[tid] = ok ? lists[e * T_TOK + r] : -1;
        wrow[tid] = ok ? wts[e * T_TOK + r]   : 0.0f;
    }
    __syncthreads();

    const float* We = ew + (size_t)e * DIM * DIM;  // [n, d] row-major

    const int wave  = tid >> 5;
    const int lane  = tid & 31;
    const int mbase = (wave & 1) * 16;             // 0 / 16
    const int nwv   = (wave >> 1) * 32;            // 0 / 32 / 64 / 96

    // ---- staging assignments -------------------------------------------
    const int arow = tid >> 3;                     // 0..31
    const int acol = (tid & 7) * 4;                // 0..28
    const int brow = tid >> 1;                     // 0..127
    const int bk   = (tid & 1) * 16;               // 0 / 16

    const int atokr = toks[arow];
    const int atokc = (atokr < 0) ? 0 : atokr;     // clamp: garbage rows never stored
    const float* __restrict__ ap = x  + (size_t)atokc * DIM + acol;
    const float* __restrict__ bp = We + (size_t)(nb + brow) * DIM + bk;

    // ---- fragment read positions (ISA 16-bit A/B layout) ----------------
    const int fm   = mbase + (lane & 15);
    const int fn0  = nwv + (lane & 15);
    const int fn1  = fn0 + 16;
    const int koff = (lane >> 4) * 8;              // lanes 16-31: K+8

    v8f cacc[2] = {{}, {}};

    // ---- stage K-step 0 into buffer 0 -----------------------------------
    {
        const float4 va = *(const float4*)ap;
        float4 vb[4];
#pragma unroll
        for (int j = 0; j < 4; ++j) vb[j] = *(const float4*)(bp + 4 * j);

        bf16 h[4], l[4];
        f32_split(va.x, h[0], l[0]); f32_split(va.y, h[1], l[1]);
        f32_split(va.z, h[2], l[2]); f32_split(va.w, h[3], l[3]);
        *(v4bf*)&Ahi[0][arow * LDK + acol] = (v4bf){h[0], h[1], h[2], h[3]};
        *(v4bf*)&Alo[0][arow * LDK + acol] = (v4bf){l[0], l[1], l[2], l[3]};

        bf16 bh[16], bl[16];
#pragma unroll
        for (int j = 0; j < 4; ++j) {
            f32_split(vb[j].x, bh[4*j+0], bl[4*j+0]);
            f32_split(vb[j].y, bh[4*j+1], bl[4*j+1]);
            f32_split(vb[j].z, bh[4*j+2], bl[4*j+2]);
            f32_split(vb[j].w, bh[4*j+3], bl[4*j+3]);
        }
        *(v8bf*)&Bhi[0][brow * LDK + bk]     = (v8bf){bh[0],bh[1],bh[2],bh[3],bh[4],bh[5],bh[6],bh[7]};
        *(v8bf*)&Bhi[0][brow * LDK + bk + 8] = (v8bf){bh[8],bh[9],bh[10],bh[11],bh[12],bh[13],bh[14],bh[15]};
        *(v8bf*)&Blo[0][brow * LDK + bk]     = (v8bf){bl[0],bl[1],bl[2],bl[3],bl[4],bl[5],bl[6],bl[7]};
        *(v8bf*)&Blo[0][brow * LDK + bk + 8] = (v8bf){bl[8],bl[9],bl[10],bl[11],bl[12],bl[13],bl[14],bl[15]};
    }
    __syncthreads();

    // ---- main loop: compute from buf, prefetch+stage next into buf^1 ----
    for (int it = 0; it < NSTEP; ++it) {
        const int  buf  = it & 1;
        const bool more = (it + 1 < NSTEP);
        const int  knxt = (it + 1) * KT;

        // issue next-slab global loads early (latency hidden behind WMMAs)
        float4 va = make_float4(0.f, 0.f, 0.f, 0.f);
        float4 vb[4];
        if (more) {
            va = *(const float4*)(ap + knxt);
#pragma unroll
            for (int j = 0; j < 4; ++j) vb[j] = *(const float4*)(bp + knxt + 4 * j);
        }

        // fragments from current buffer
        const int aoff = fm * LDK + koff;
        const v8bf a0h = *(const v8bf*)&Ahi[buf][aoff];
        const v8bf a1h = *(const v8bf*)&Ahi[buf][aoff + 16];
        const v8bf a0l = *(const v8bf*)&Alo[buf][aoff];
        const v8bf a1l = *(const v8bf*)&Alo[buf][aoff + 16];
        const v16bf ah = __builtin_shufflevector(a0h, a1h, 0,1,2,3,4,5,6,7,8,9,10,11,12,13,14,15);
        const v16bf al = __builtin_shufflevector(a0l, a1l, 0,1,2,3,4,5,6,7,8,9,10,11,12,13,14,15);

        const int b0off = fn0 * LDK + koff;
        const v8bf b00h = *(const v8bf*)&Bhi[buf][b0off];
        const v8bf b01h = *(const v8bf*)&Bhi[buf][b0off + 16];
        const v8bf b00l = *(const v8bf*)&Blo[buf][b0off];
        const v8bf b01l = *(const v8bf*)&Blo[buf][b0off + 16];
        const v16bf bh0 = __builtin_shufflevector(b00h, b01h, 0,1,2,3,4,5,6,7,8,9,10,11,12,13,14,15);
        const v16bf bl0 = __builtin_shufflevector(b00l, b01l, 0,1,2,3,4,5,6,7,8,9,10,11,12,13,14,15);

        const int b1off = fn1 * LDK + koff;
        const v8bf b10h = *(const v8bf*)&Bhi[buf][b1off];
        const v8bf b11h = *(const v8bf*)&Bhi[buf][b1off + 16];
        const v8bf b10l = *(const v8bf*)&Blo[buf][b1off];
        const v8bf b11l = *(const v8bf*)&Blo[buf][b1off + 16];
        const v16bf bh1 = __builtin_shufflevector(b10h, b11h, 0,1,2,3,4,5,6,7,8,9,10,11,12,13,14,15);
        const v16bf bl1 = __builtin_shufflevector(b10l, b11l, 0,1,2,3,4,5,6,7,8,9,10,11,12,13,14,15);

        // hi*hi + hi*lo + lo*hi (f32 accumulate): near-fp32 accuracy
        cacc[0] = __builtin_amdgcn_wmma_f32_16x16x32_bf16(false, ah, false, bh0, (short)0, cacc[0], false, false);
        cacc[0] = __builtin_amdgcn_wmma_f32_16x16x32_bf16(false, ah, false, bl0, (short)0, cacc[0], false, false);
        cacc[0] = __builtin_amdgcn_wmma_f32_16x16x32_bf16(false, al, false, bh0, (short)0, cacc[0], false, false);
        cacc[1] = __builtin_amdgcn_wmma_f32_16x16x32_bf16(false, ah, false, bh1, (short)0, cacc[1], false, false);
        cacc[1] = __builtin_amdgcn_wmma_f32_16x16x32_bf16(false, ah, false, bl1, (short)0, cacc[1], false, false);
        cacc[1] = __builtin_amdgcn_wmma_f32_16x16x32_bf16(false, al, false, bh1, (short)0, cacc[1], false, false);

        // convert + store next slab into the other buffer
        if (more) {
            const int nbuf = buf ^ 1;
            bf16 h[4], l[4];
            f32_split(va.x, h[0], l[0]); f32_split(va.y, h[1], l[1]);
            f32_split(va.z, h[2], l[2]); f32_split(va.w, h[3], l[3]);
            *(v4bf*)&Ahi[nbuf][arow * LDK + acol] = (v4bf){h[0], h[1], h[2], h[3]};
            *(v4bf*)&Alo[nbuf][arow * LDK + acol] = (v4bf){l[0], l[1], l[2], l[3]};

            bf16 bh[16], bl[16];
#pragma unroll
            for (int j = 0; j < 4; ++j) {
                f32_split(vb[j].x, bh[4*j+0], bl[4*j+0]);
                f32_split(vb[j].y, bh[4*j+1], bl[4*j+1]);
                f32_split(vb[j].z, bh[4*j+2], bl[4*j+2]);
                f32_split(vb[j].w, bh[4*j+3], bl[4*j+3]);
            }
            *(v8bf*)&Bhi[nbuf][brow * LDK + bk]     = (v8bf){bh[0],bh[1],bh[2],bh[3],bh[4],bh[5],bh[6],bh[7]};
            *(v8bf*)&Bhi[nbuf][brow * LDK + bk + 8] = (v8bf){bh[8],bh[9],bh[10],bh[11],bh[12],bh[13],bh[14],bh[15]};
            *(v8bf*)&Blo[nbuf][brow * LDK + bk]     = (v8bf){bl[0],bl[1],bl[2],bl[3],bl[4],bl[5],bl[6],bl[7]};
            *(v8bf*)&Blo[nbuf][brow * LDK + bk + 8] = (v8bf){bl[8],bl[9],bl[10],bl[11],bl[12],bl[13],bl[14],bl[15]};
        }
        __syncthreads();
    }

    // ---- epilogue: D layout: VGPR r -> row mbase + r + 8*(lane>=16) -----
#pragma unroll
    for (int half = 0; half < 2; ++half) {
        const int ncol = nb + nwv + half * 16 + (lane & 15);
#pragma unroll
        for (int r = 0; r < 8; ++r) {
            const int m   = mbase + r + ((lane >> 4) * 8);
            const int tok = toks[m];
            if (tok >= 0) {
                const float val = cacc[half][r] * wrow[m];
                float* p = out + (size_t)tok * DIM + ncol;
                if (ACC) *p = *p + val;
                else     *p = val;
            }
        }
    }
}

// ---------------------------------------------------------------------------
// Launch. Inputs: x [4,2048,1024] f32, gate_w [8,1024] f32, expert_w
// [8,1024,1024] f32. Output: [4,2048,1024] f32.  Workspace ~1.05 MB.
// ---------------------------------------------------------------------------
extern "C" void kernel_launch(void* const* d_in, const int* in_sizes, int n_in,
                              void* d_out, int out_size, void* d_ws, size_t ws_size,
                              hipStream_t stream)
{
    (void)in_sizes; (void)n_in; (void)out_size; (void)ws_size;
    const float* x  = (const float*)d_in[0];
    const float* gw = (const float*)d_in[1];
    const float* ew = (const float*)d_in[2];
    float* out = (float*)d_out;

    char* ws = (char*)d_ws;
    const size_t ET = (size_t)NEXP * T_TOK;          // 65536 entries
    int*   c0 = (int*)(ws + 0);
    int*   c1 = (int*)(ws + 32);
    int*   l0 = (int*)(ws + 256);
    float* w0 = (float*)(ws + 256 + ET * 4);
    int*   l1 = (int*)(ws + 256 + ET * 8);
    float* w1 = (float*)(ws + 256 + ET * 12);

    hipMemsetAsync(ws, 0, 64, stream);               // zero atomic counters

    moe_gate<<<T_TOK / 8, 256, 0, stream>>>(x, gw, c0, c1, l0, w0, l1, w1);

    dim3 grid(DIM / NT, T_TOK / MT, NEXP);           // (8, 256, 8)
    // Pass A: rank-0 experts — every token has exactly one entry: full store.
    moe_gemm<0><<<grid, 256, 0, stream>>>(x, ew, c0, l0, w0, out);
    // Pass B: rank-1 experts — each token exactly once: race-free accumulate.
    moe_gemm<1><<<grid, 256, 0, stream>>>(x, ew, c1, l1, w1, out);
}
